// GSAPredict_58480274702470
// MI455X (gfx1250) — matrix operand order repaired and checked by reading, not compiled, not run on hardware
//
#include <hip/hip_runtime.h>
#include <hip/hip_bf16.h>
#include <math.h>

// ---------------- problem constants (match reference) ----------------
#define B_N     8
#define H_N     8
#define DM_N    512
#define DK_N    64
#define S_TOT   6144          // T + K_STEPS
#define WN      6113          // k + T - M + 1
#define M_N     32
#define VROWS   6112          // WN - 1
#define JT_N    383           // ceil(WN/16)

typedef __bf16 bf16_t;
typedef __attribute__((ext_vector_type(16))) __bf16 v16bf;
typedef __attribute__((ext_vector_type(8)))  __bf16 v8bf;
typedef __attribute__((ext_vector_type(8)))  float  v8f;

// A-fragment for V_WMMA_F32_16X16X32_BF16:
// lanes 0-15 : row M=lane,   elems[0..7]=K(kk+0..7),  elems[8..15]=K(kk+16..23)
// lanes 16-31: row M=lane-16,elems[0..7]=K(kk+8..15), elems[8..15]=K(kk+24..31)
// => per lane: two contiguous 16B loads at (base + half*8) and (+16 elems).
static __device__ __forceinline__ v16bf load_afrag(const bf16_t* p) {
  v8bf lo = *(const v8bf*)(p);
  v8bf hi = *(const v8bf*)(p + 16);
  return __builtin_shufflevector(lo, hi, 0,1,2,3,4,5,6,7,8,9,10,11,12,13,14,15);
}

static __device__ __forceinline__ v8f wmma_bf16(v16bf a, v16bf b, v8f c) {
  return __builtin_amdgcn_wmma_f32_16x16x32_bf16(false, a, false, b, (short)0, c,
                                                 false, false);
}

// ---------------- data prep ----------------
__global__ void build_xbf_kernel(const float* __restrict__ memory,
                                 const float* __restrict__ x_pre,
                                 bf16_t* __restrict__ xbf) {
  size_t idx = (size_t)blockIdx.x * blockDim.x + threadIdx.x;
  size_t total = (size_t)B_N * S_TOT * DM_N;
  if (idx >= total) return;
  int d = (int)(idx % DM_N);
  size_t sb = idx / DM_N;
  int s = (int)(sb % S_TOT);
  int b = (int)(sb / S_TOT);
  float v = (s < S_TOT - 1) ? memory[((size_t)b * (S_TOT - 1) + s) * DM_N + d]
                            : x_pre[(size_t)b * DM_N + d];
  xbf[idx] = (bf16_t)v;
}

__global__ void f32_to_bf16_kernel(const float* __restrict__ src,
                                   bf16_t* __restrict__ dst, size_t n) {
  size_t i = (size_t)blockIdx.x * blockDim.x + threadIdx.x;
  if (i < n) dst[i] = (bf16_t)src[i];
}

// Pack weight W (N x Kd, row-major, out = X @ W^T) into WMMA B-fragment order.
// Fragment (ntile, kc) is 512 bf16, laid out [lane][16 elems] contiguous.
// B 32x16 layout matches A-fragment K order: lanes 0-15 K 0..7,16..23; lanes 16-31 K 8..15,24..31.
// => element e of lane's v16bf is K = (e&7) + (lane>=16)*8 + (e>=8)*16, col = n0 + (lane&15).
__global__ void pack_w_kernel(const float* __restrict__ W,
                              const int* __restrict__ Gm,
                              bf16_t* __restrict__ out,
                              int Kd, int KC, size_t total) {
  size_t p = (size_t)blockIdx.x * blockDim.x + threadIdx.x;
  if (p >= total) return;
  int blk   = KC * 512;
  int ntile = (int)(p / blk);
  int rem   = (int)(p % blk);
  int kc    = rem / 512;
  int q     = rem % 512;
  int lane  = q >> 4, e = q & 15;
  int n  = ntile * 16 + (lane & 15);
  int kl = (e & 7) + ((lane >> 4) << 3) + ((e >> 3) << 4);
  int d  = kc * 32 + kl;
  float v = W[(size_t)n * Kd + d];
  if (Gm) v *= (float)Gm[(n & 63) * 64 + (d & 63)];
  out[p] = (bf16_t)v;
}

// ---------------- generic projection GEMM + optional per-head l2-norm ----------
// One wave computes (16*ROWT) rows x HD cols for one (batch, head).
// ROWT=2 amortizes each B-fragment load over 2 WMMAs and doubles MMA ILP.
// grid: x = ceil(R/(16*ROWT)/4), y = 8 heads, z = B.  block = 128 (4 waves).
template <int HD, int KC, int DO_NORM, int ROWT>
__global__ void __launch_bounds__(128)
gemm_norm_t(const bf16_t* __restrict__ X, long bsX, int r0, int rowClamp,
            const bf16_t* __restrict__ Wpack, const float* __restrict__ bias,
            int R, bf16_t* __restrict__ outB, float* __restrict__ outF,
            long bsO) {
  constexpr int NT = HD / 16;
  constexpr int Kd = KC * 32;
  constexpr int N  = HD * 8;

  int wid = threadIdx.x >> 5, lane = threadIdx.x & 31;
  int pairIdx = blockIdx.x * 4 + wid;
  int rtBase = pairIdx * ROWT;                 // in 16-row tile units
  if (rtBase * 16 >= R) return;
  int head = blockIdx.y, b = blockIdx.z;
  int n0 = head * HD;

  v8f acc[ROWT][NT];
#pragma unroll
  for (int rw = 0; rw < ROWT; ++rw)
#pragma unroll
    for (int t = 0; t < NT; ++t) acc[rw][t] = (v8f){};

  const bf16_t* xrow[ROWT];
#pragma unroll
  for (int rw = 0; rw < ROWT; ++rw) {
    int rowA = r0 + (rtBase + rw) * 16 + (lane & 15);
    if (rowA >= rowClamp) rowA = rowClamp - 1;
    xrow[rw] = X + (size_t)b * bsX + (size_t)rowA * Kd;
  }
  const bf16_t* wbase = Wpack + ((size_t)(n0 >> 4) * KC) * 512 + lane * 16;
  int halfOff8 = (lane >> 4) << 3;

#pragma unroll
  for (int kc = 0; kc < KC; ++kc) {
    v16bf a[ROWT];
#pragma unroll
    for (int rw = 0; rw < ROWT; ++rw)
      a[rw] = load_afrag(xrow[rw] + kc * 32 + halfOff8);
#pragma unroll
    for (int t = 0; t < NT; ++t) {
      v16bf bb = *(const v16bf*)(wbase + ((size_t)t * KC + kc) * 512);
#pragma unroll
      for (int rw = 0; rw < ROWT; ++rw)
        acc[rw][t] = wmma_bf16(a[rw], bb, acc[rw][t]);
    }
  }

  int ncol = lane & 15;
  int half8 = (lane >> 4) << 3;
#pragma unroll
  for (int rw = 0; rw < ROWT; ++rw) {
    float rs[8];
#pragma unroll
    for (int r = 0; r < 8; ++r) rs[r] = 0.0f;
#pragma unroll
    for (int t = 0; t < NT; ++t) {
      float bv = bias[n0 + t * 16 + ncol];
#pragma unroll
      for (int r = 0; r < 8; ++r) {
        float v = acc[rw][t][r] + bv;
        acc[rw][t][r] = v;
        rs[r] += v * v;
      }
    }
    if (DO_NORM) {
      // reduce row sum-of-squares across the 16 lanes of each half
#pragma unroll
      for (int m = 1; m < 16; m <<= 1)
#pragma unroll
        for (int r = 0; r < 8; ++r) rs[r] += __shfl_xor(rs[r], m, 32);
      float inv[8];
#pragma unroll
      for (int r = 0; r < 8; ++r) inv[r] = 1.0f / fmaxf(sqrtf(rs[r]), 1e-12f);
#pragma unroll
      for (int t = 0; t < NT; ++t)
#pragma unroll
        for (int r = 0; r < 8; ++r) {
          int rowLoc = (rtBase + rw) * 16 + r + half8;
          if (rowLoc < R)
            outB[(size_t)b * bsO + (size_t)rowLoc * N + n0 + t * 16 + ncol] =
                (bf16_t)(acc[rw][t][r] * inv[r]);
        }
    } else {
#pragma unroll
      for (int t = 0; t < NT; ++t)
#pragma unroll
        for (int r = 0; r < 8; ++r) {
          int rowLoc = (rtBase + rw) * 16 + r + half8;
          if (rowLoc < R)
            outF[(size_t)b * bsO + (size_t)rowLoc * N + n0 + t * 16 + ncol] =
                acc[rw][t][r];
        }
    }
  }
}

// ---------------- banded tn: tn[b,h,j] = mean_m Q[m].K[m+j] -------------------
// One wave per 16-wide j tile: 2 m-tiles x 3 key-tiles x 2 k-chunks of WMMA.
__global__ void __launch_bounds__(128)
band_tn_kernel(const bf16_t* __restrict__ Qhat, const bf16_t* __restrict__ Khat,
               float* __restrict__ tn) {
  __shared__ float ldsD[4][6][256];
  int wid = threadIdx.x >> 5, lane = threadIdx.x & 31;
  int jt = blockIdx.x * 4 + wid;
  int jtc = jt < JT_N ? jt : JT_N - 1;
  int h = blockIdx.y, b = blockIdx.z;
  int j0 = jtc * 16;
  int halfOff8 = (lane >> 4) << 3;
  int halfOff16 = (lane >> 4) << 4;
  const bf16_t* Qb = Qhat + ((size_t)b * 32) * DM_N + h * DK_N;
  const bf16_t* Kb = Khat + ((size_t)b * S_TOT) * DM_N + h * DK_N;

#pragma unroll
  for (int mt = 0; mt < 2; ++mt) {
    int m = mt * 16 + (lane & 15);
#pragma unroll
    for (int kt = 0; kt < 3; ++kt) {
      int skey = j0 + kt * 16 + (lane & 15);
      if (skey > S_TOT - 1) skey = S_TOT - 1;
      v8f acc = {};
#pragma unroll
      for (int kc = 0; kc < 2; ++kc) {
        v16bf a = load_afrag(Qb + (size_t)m * DM_N + kc * 32 + halfOff8);
        v16bf bb =
            load_afrag(Kb + (size_t)skey * DM_N + kc * 32 + halfOff8);
        acc = wmma_bf16(a, bb, acc);
      }
      int n = lane & 15, hr = (lane >> 4) << 3;
#pragma unroll
      for (int r = 0; r < 8; ++r)
        ldsD[wid][mt * 3 + kt][(r + hr) * 16 + n] = acc[r];
    }
  }
  __syncthreads();
  if (lane < 16) {
    int jj = lane;
    float s = 0.0f;
#pragma unroll 4
    for (int m = 0; m < 32; ++m) {
      int c = m + jj;
      s += ldsD[wid][(m >> 4) * 3 + (c >> 4)][(m & 15) * 16 + (c & 15)];
    }
    int j = jt * 16 + jj;
    if (jt < JT_N && j < WN)
      tn[((size_t)(b * H_N + h)) * WN + j] = s * (1.0f / 32.0f);
  }
}

// ---------------- small dots (aux / pos) --------------------------------------
__global__ void dot_small_kernel(const bf16_t* __restrict__ Q1,
                                 const bf16_t* __restrict__ Khat, int C, int Nld,
                                 float* __restrict__ out) {
  int h = blockIdx.y, b = blockIdx.z;
  int j = blockIdx.x * blockDim.x + threadIdx.x;
  if (j >= WN) return;
  const bf16_t* q = Q1 + ((size_t)b * 16 + 15) * Nld + h * C;
  int row = (S_TOT - WN) + j;
  const bf16_t* kr = Khat + ((size_t)b * S_TOT + row) * Nld + h * C;
  float acc = 0.0f;
  for (int c = 0; c < C; ++c) acc += (float)q[c] * (float)kr[c];
  out[((size_t)(b * H_N + h)) * WN + j] = acc;
}

// ---------------- softmax over Wn ---------------------------------------------
__global__ void softmax_kernel(const float* __restrict__ tn,
                               const float* __restrict__ d2,
                               const float* __restrict__ d3,
                               const float* __restrict__ w0p,
                               const float* __restrict__ w1p,
                               const float* __restrict__ w2p,
                               float* __restrict__ attn) {
  __shared__ float red[256];
  int tid = threadIdx.x;
  size_t base = (size_t)blockIdx.x * WN;
  float w0 = w0p[0], w1 = w1p[0], w2 = w2p[0];
  float mx = -3.4e38f;
  for (int j = tid; j < WN; j += 256) {
    float l = w0 * tn[base + j] + w1 * d2[base + j] + w2 * d3[base + j];
    mx = fmaxf(mx, l);
  }
  red[tid] = mx;
  __syncthreads();
  for (int s = 128; s > 0; s >>= 1) {
    if (tid < s) red[tid] = fmaxf(red[tid], red[tid + s]);
    __syncthreads();
  }
  mx = red[0];
  __syncthreads();
  float sum = 0.0f;
  for (int j = tid; j < WN; j += 256) {
    float l = w0 * tn[base + j] + w1 * d2[base + j] + w2 * d3[base + j];
    float e = expf(l - mx);
    attn[base + j] = e;
    sum += e;
  }
  red[tid] = sum;
  __syncthreads();
  for (int s = 128; s > 0; s >>= 1) {
    if (tid < s) red[tid] += red[tid + s];
    __syncthreads();
  }
  float inv = 1.0f / red[0];
  for (int j = tid; j < WN; j += 256) attn[base + j] *= inv;
}

// ---------------- hist = attn[:, :Wn-1] . V -----------------------------------
__global__ void hist_kernel(const float* __restrict__ attn,
                            const float* __restrict__ Vf,
                            float* __restrict__ hist) {
  int h = blockIdx.x, b = blockIdx.y, dk = threadIdx.x;
  const float* ar = attn + (size_t)(b * H_N + h) * WN;
  const float* vcol = Vf + (size_t)b * VROWS * DM_N + h * DK_N + dk;
  float acc = 0.0f;
  for (int j = 0; j < WN - 1; ++j) {
    __builtin_prefetch(vcol + (size_t)(j + 16) * DM_N, 0, 0);
    acc += ar[j] * vcol[(size_t)j * DM_N];
  }
  hist[(size_t)(b * H_N + h) * DK_N + dk] = acc;
}

// ---------------- 31-step GRU over recent rows --------------------------------
__global__ void gru_kernel(const float* __restrict__ memory,
                           const float* __restrict__ W_ih,
                           const float* __restrict__ W_hh,
                           const float* __restrict__ b_ih,
                           const float* __restrict__ b_hh,
                           float* __restrict__ res) {
  __shared__ float xs[DM_N];
  __shared__ float hb[DK_N];
  int b = blockIdx.x, d = threadIdx.x;
  hb[d] = 0.0f;
  __syncthreads();
  for (int t = 0; t < M_N - 1; ++t) {
    int row = S_TOT - M_N + t;  // memory row (all < S-1)
    for (int i = d; i < DM_N; i += DK_N)
      xs[i] = memory[((size_t)b * (S_TOT - 1) + row) * DM_N + i];
    __syncthreads();
    float gir = b_ih[d], giz = b_ih[64 + d], gin = b_ih[128 + d];
    const float* wr = W_ih + (size_t)d * DM_N;
    const float* wz = W_ih + (size_t)(64 + d) * DM_N;
    const float* wn = W_ih + (size_t)(128 + d) * DM_N;
    for (int dd = 0; dd < DM_N; ++dd) {
      float x = xs[dd];
      gir += x * wr[dd]; giz += x * wz[dd]; gin += x * wn[dd];
    }
    float ghr = b_hh[d], ghz = b_hh[64 + d], ghn = b_hh[128 + d];
    const float* ur = W_hh + (size_t)d * DK_N;
    const float* uz = W_hh + (size_t)(64 + d) * DK_N;
    const float* un = W_hh + (size_t)(128 + d) * DK_N;
    for (int dd = 0; dd < DK_N; ++dd) {
      float hv = hb[dd];
      ghr += hv * ur[dd]; ghz += hv * uz[dd]; ghn += hv * un[dd];
    }
    float r = 1.0f / (1.0f + expf(-(gir + ghr)));
    float z = 1.0f / (1.0f + expf(-(giz + ghz)));
    float nst = tanhf(gin + r * ghn);
    float hnew = (1.0f - z) * nst + z * hb[d];
    __syncthreads();
    hb[d] = hnew;
    __syncthreads();
  }
  res[(size_t)b * DK_N + d] = hb[d];
}

// ---------------- output: x_pre + (deta @ (WO*mask)^T + bO) -------------------
__global__ void final_kernel(const float* __restrict__ x_pre,
                             const float* __restrict__ hist,
                             const float* __restrict__ attn,
                             const float* __restrict__ res,
                             const float* __restrict__ WO,
                             const float* __restrict__ bO,
                             const int* __restrict__ G,
                             float* __restrict__ out) {
  __shared__ float detas[DM_N];
  int b = blockIdx.x, d = threadIdx.x;  // 512 threads
  int h = d >> 6, dk = d & 63;
  float a_last = attn[(size_t)(b * H_N + h) * WN + (WN - 1)];
  detas[d] = hist[(size_t)(b * H_N + h) * DK_N + dk] +
             a_last * res[(size_t)b * DK_N + dk];
  __syncthreads();
  float acc = bO[d];
  const float* wrow = WO + (size_t)d * DM_N;
  const int* grow = G + (size_t)(d & 63) * 64;
  for (int e = 0; e < DM_N; ++e)
    acc += detas[e] * wrow[e] * (float)grow[e & 63];
  out[(size_t)b * DM_N + d] = x_pre[(size_t)b * DM_N + d] + acc;
}

// ---------------- host side ----------------------------------------------------
extern "C" void kernel_launch(void* const* d_in, const int* in_sizes, int n_in,
                              void* d_out, int out_size, void* d_ws,
                              size_t ws_size, hipStream_t stream) {
  (void)in_sizes; (void)n_in; (void)out_size; (void)ws_size;
  const float* memory = (const float*)d_in[0];
  const float* x_pre  = (const float*)d_in[1];
  const float* aux    = (const float*)d_in[2];
  const float* pos    = (const float*)d_in[3];
  const float* Wq  = (const float*)d_in[4];   const float* bq  = (const float*)d_in[5];
  const float* Wk  = (const float*)d_in[6];   const float* bk  = (const float*)d_in[7];
  const float* Wv  = (const float*)d_in[8];   const float* bv  = (const float*)d_in[9];
  const float* Wqa = (const float*)d_in[10];  const float* bqa = (const float*)d_in[11];
  const float* Wka = (const float*)d_in[12];  const float* bka = (const float*)d_in[13];
  const float* Wqp = (const float*)d_in[14];  const float* bqp = (const float*)d_in[15];
  const float* Wkp = (const float*)d_in[16];  const float* bkp = (const float*)d_in[17];
  const float* W_ih = (const float*)d_in[18]; const float* W_hh = (const float*)d_in[19];
  const float* b_ih = (const float*)d_in[20]; const float* b_hh = (const float*)d_in[21];
  const float* WO = (const float*)d_in[22];   const float* bO = (const float*)d_in[23];
  const float* w   = (const float*)d_in[24];
  const float* w_a = (const float*)d_in[25];
  const float* w_p = (const float*)d_in[26];
  const int*   G   = (const int*)d_in[27];
  float* out = (float*)d_out;

  char* ws = (char*)d_ws;
  size_t off = 0;
  auto alloc = [&](size_t bytes) -> char* {
    char* p = ws + off;
    off = (off + bytes + 255) & ~(size_t)255;
    return p;
  };

  bf16_t* xbf   = (bf16_t*)alloc((size_t)B_N * S_TOT * DM_N * 2);
  bf16_t* auxbf = (bf16_t*)alloc((size_t)B_N * S_TOT * 256 * 2);
  bf16_t* posbf = (bf16_t*)alloc((size_t)B_N * S_TOT * 128 * 2);
  bf16_t* wqP  = (bf16_t*)alloc((size_t)512 * 512 * 2);
  bf16_t* wkP  = (bf16_t*)alloc((size_t)512 * 512 * 2);
  bf16_t* wvP  = (bf16_t*)alloc((size_t)512 * 512 * 2);
  bf16_t* wqaP = (bf16_t*)alloc((size_t)256 * 256 * 2);
  bf16_t* wkaP = (bf16_t*)alloc((size_t)256 * 256 * 2);
  bf16_t* wqpP = (bf16_t*)alloc((size_t)128 * 128 * 2);
  bf16_t* wkpP = (bf16_t*)alloc((size_t)128 * 128 * 2);
  bf16_t* khat  = (bf16_t*)alloc((size_t)B_N * S_TOT * 512 * 2);
  bf16_t* qhat  = (bf16_t*)alloc((size_t)B_N * 32 * 512 * 2);
  bf16_t* kahat = (bf16_t*)alloc((size_t)B_N * S_TOT * 256 * 2);
  bf16_t* qa1   = (bf16_t*)alloc((size_t)B_N * 16 * 256 * 2);
  bf16_t* kphat = (bf16_t*)alloc((size_t)B_N * S_TOT * 128 * 2);
  bf16_t* qp1   = (bf16_t*)alloc((size_t)B_N * 16 * 128 * 2);
  float* vf   = (float*)alloc((size_t)B_N * VROWS * 512 * 4);
  float* tnb  = (float*)alloc((size_t)B_N * H_N * WN * 4);
  float* d2b  = (float*)alloc((size_t)B_N * H_N * WN * 4);
  float* d3b  = (float*)alloc((size_t)B_N * H_N * WN * 4);
  float* attn = (float*)alloc((size_t)B_N * H_N * WN * 4);
  float* histb = (float*)alloc((size_t)B_N * 512 * 4);
  float* resb  = (float*)alloc((size_t)B_N * 64 * 4);

  // 1) conversions
  {
    size_t n = (size_t)B_N * S_TOT * DM_N;
    build_xbf_kernel<<<dim3((unsigned)((n + 255) / 256)), dim3(256), 0, stream>>>(
        memory, x_pre, xbf);
    size_t na = (size_t)B_N * S_TOT * 256;
    f32_to_bf16_kernel<<<dim3((unsigned)((na + 255) / 256)), dim3(256), 0,
                         stream>>>(aux, auxbf, na);
    size_t np = (size_t)B_N * S_TOT * 128;
    f32_to_bf16_kernel<<<dim3((unsigned)((np + 255) / 256)), dim3(256), 0,
                         stream>>>(pos, posbf, np);
  }
  // 2) weight packing
  {
    size_t t512 = (size_t)512 * 512, t256 = (size_t)256 * 256, t128 = (size_t)128 * 128;
    pack_w_kernel<<<dim3((unsigned)((t512 + 255) / 256)), dim3(256), 0, stream>>>(Wq, G, wqP, 512, 16, t512);
    pack_w_kernel<<<dim3((unsigned)((t512 + 255) / 256)), dim3(256), 0, stream>>>(Wk, G, wkP, 512, 16, t512);
    pack_w_kernel<<<dim3((unsigned)((t512 + 255) / 256)), dim3(256), 0, stream>>>(Wv, G, wvP, 512, 16, t512);
    pack_w_kernel<<<dim3((unsigned)((t256 + 255) / 256)), dim3(256), 0, stream>>>(Wqa, nullptr, wqaP, 256, 8, t256);
    pack_w_kernel<<<dim3((unsigned)((t256 + 255) / 256)), dim3(256), 0, stream>>>(Wka, nullptr, wkaP, 256, 8, t256);
    pack_w_kernel<<<dim3((unsigned)((t128 + 255) / 256)), dim3(256), 0, stream>>>(Wqp, nullptr, wqpP, 128, 4, t128);
    pack_w_kernel<<<dim3((unsigned)((t128 + 255) / 256)), dim3(256), 0, stream>>>(Wkp, nullptr, wkpP, 128, 4, t128);
  }
  // 3) projections (WMMA GEMMs), all loops compile-time unrolled
  long bsX512 = (long)S_TOT * 512, bsX256 = (long)S_TOT * 256, bsX128 = (long)S_TOT * 128;
  // K-hat: all S rows, normalized.  6144 rows / 32 per wave / 4 waves = 48 blocks
  gemm_norm_t<64, 16, 1, 2><<<dim3(48, 8, 8), dim3(128), 0, stream>>>(
      xbf, bsX512, 0, S_TOT, wkP, bk, S_TOT, khat, nullptr, (long)S_TOT * 512);
  // Q-hat: last 32 rows (one wave covers both tiles)
  gemm_norm_t<64, 16, 1, 2><<<dim3(1, 8, 8), dim3(128), 0, stream>>>(
      xbf, bsX512, S_TOT - 32, S_TOT, wqP, bq, 32, qhat, nullptr, (long)32 * 512);
  // V: rows [S-Wn, S-1), raw f32. 6112 rows -> 191 pairs -> 48 blocks
  gemm_norm_t<64, 16, 0, 2><<<dim3(48, 8, 8), dim3(128), 0, stream>>>(
      xbf, bsX512, S_TOT - WN, S_TOT, wvP, bv, VROWS, nullptr, vf,
      (long)VROWS * 512);
  // KA-hat / QA-hat
  gemm_norm_t<32, 8, 1, 2><<<dim3(48, 8, 8), dim3(128), 0, stream>>>(
      auxbf, bsX256, 0, S_TOT, wkaP, bka, S_TOT, kahat, nullptr,
      (long)S_TOT * 256);
  gemm_norm_t<32, 8, 1, 1><<<dim3(1, 8, 8), dim3(128), 0, stream>>>(
      auxbf, bsX256, S_TOT - 16, S_TOT, wqaP, bqa, 16, qa1, nullptr,
      (long)16 * 256);
  // KP-hat / QP-hat
  gemm_norm_t<16, 4, 1, 2><<<dim3(48, 8, 8), dim3(128), 0, stream>>>(
      posbf, bsX128, 0, S_TOT, wkpP, bkp, S_TOT, kphat, nullptr,
      (long)S_TOT * 128);
  gemm_norm_t<16, 4, 1, 1><<<dim3(1, 8, 8), dim3(128), 0, stream>>>(
      posbf, bsX128, S_TOT - 16, S_TOT, wqpP, bqp, 16, qp1, nullptr,
      (long)16 * 128);
  // 4) banded tn (WMMA)
  band_tn_kernel<<<dim3((JT_N + 3) / 4, 8, 8), dim3(128), 0, stream>>>(qhat, khat, tnb);
  // 5) aux/pos dots
  dot_small_kernel<<<dim3((WN + 255) / 256, 8, 8), dim3(256), 0, stream>>>(qa1, kahat, 32, 256, d2b);
  dot_small_kernel<<<dim3((WN + 255) / 256, 8, 8), dim3(256), 0, stream>>>(qp1, kphat, 16, 128, d3b);
  // 6) softmax
  softmax_kernel<<<dim3(B_N * H_N), dim3(256), 0, stream>>>(tnb, d2b, d3b, w, w_a, w_p, attn);
  // 7) hist
  hist_kernel<<<dim3(H_N, B_N), dim3(64), 0, stream>>>(attn, vf, histb);
  // 8) GRU
  gru_kernel<<<dim3(B_N), dim3(64), 0, stream>>>(memory, W_ih, W_hh, b_ih, b_hh, resb);
  // 9) output
  final_kernel<<<dim3(B_N), dim3(512), 0, stream>>>(x_pre, histb, attn, resb, WO, bO, G, out);
}
